// GraphMAE_42236708389386
// MI455X (gfx1250) — matrix operand order repaired and checked by reading, not compiled, not run on hardware
//
#include <hip/hip_runtime.h>
#include <hip/hip_bf16.h>

// Problem constants (match reference).
#define NNODES  50000
#define DIN     64
#define HIDDEN  128
#define NHEADS  4
#define CH      32
#define NEG_SLOPE 0.2f

typedef __attribute__((ext_vector_type(16))) __bf16 v16bf;
typedef __attribute__((ext_vector_type(8)))  float  v8f;

// ---------------------------------------------------------------------------
// Order-preserving float<->uint encoding for atomicMax-based segment max.
// ---------------------------------------------------------------------------
__device__ __forceinline__ unsigned encOrd(float f) {
  unsigned u = __float_as_uint(f);
  return (u & 0x80000000u) ? ~u : (u | 0x80000000u);
}
__device__ __forceinline__ float decOrd(unsigned u) {
  return __uint_as_float((u & 0x80000000u) ? (u & 0x7FFFFFFFu) : ~u);
}
#define ENC_NEG_INF 0x007FFFFFu  // encOrd(-inf)

// ---------------------------------------------------------------------------
// Weight repack: f32 [K x Ncols] row-major -> bf16 swizzled to this file's contiguous-k
// 16-bit B lane layout (k = 16*half + i, not the ISA table order). Output group index:
//   ((ntile*KSTEPS + ks)*32 + lane) -> 16 contiguous bf16 (32B, aligned)
// where lane<16 covers K=k0..k0+15 at column ntile*16+lane,
//       lane>=16 covers K=k0+16..k0+31 at column ntile*16+lane-16.
// ---------------------------------------------------------------------------
__global__ void k_repack_w(const float* __restrict__ W, __bf16* __restrict__ out,
                           int K, int Ncols) {
  const int ksteps = K / 32, ntiles = Ncols / 16;
  int t = blockIdx.x * blockDim.x + threadIdx.x;
  if (t >= ntiles * ksteps * 32) return;
  const int lane = t & 31;
  const int ks   = (t >> 5) % ksteps;
  const int nt   = (t >> 5) / ksteps;
  const int half = lane >> 4, l16 = lane & 15;
  const float* __restrict__ src =
      W + (size_t)(ks * 32 + half * 16) * Ncols + nt * 16 + l16;
  v16bf v;
#pragma unroll
  for (int i = 0; i < 16; ++i) v[i] = (__bf16)src[(size_t)i * Ncols];
  *((v16bf*)(out + (size_t)t * 16)) = v;
}

// ---------------------------------------------------------------------------
// WMMA bf16 GEMM: C[M,Ncols] = act( A[M,K] @ W + bias ), W pre-repacked bf16.
// One wave per 16x16 tile, K fully unrolled (K/32 back-to-back WMMAs).
// A loaded as 4x float4 per lane per step (two contiguous 8-float runs
// in k = 16*half + s order, as in repacked B), converted to bf16 in-register.
// blockDim.x = 32*(Ncols/16); gridDim.x = M/16.
// ---------------------------------------------------------------------------
template <int K, bool HASBIAS, bool RELU, bool HASMASK>
__global__ void k_gemm_bf16(const float* __restrict__ A, const __bf16* __restrict__ Wr,
                            const float* __restrict__ bias, const int* __restrict__ mask,
                            float* __restrict__ C, int M, int Ncols) {
  constexpr int KSTEPS = K / 32;
  const int lane = threadIdx.x & 31;
  const int wave = threadIdx.x >> 5;
  const int half = lane >> 4;   // 0: lanes 0-15, 1: lanes 16-31
  const int l16  = lane & 15;
  const int m0 = blockIdx.x * 16;
  const int n0 = wave * 16;
  if (m0 >= M) return;          // uniform (exact tiling: N % 16 == 0)

  const int row = m0 + l16;
  float ascale = 1.0f;
  if (HASMASK) ascale = mask[row] ? 0.0f : 1.0f;
  const float* __restrict__ arow = A + (size_t)row * K;
  const v16bf* __restrict__ bvp = (const v16bf*)Wr + (size_t)wave * KSTEPS * 32 + lane;

  v8f acc = {};
#pragma unroll
  for (int ks = 0; ks < KSTEPS; ++ks) {
    const int k0 = ks * 32;
    const float4* __restrict__ a0 = (const float4*)(arow + k0 + half * 16);
    const float4* __restrict__ a1 = (const float4*)(arow + k0 + half * 16 + 8);
    const float4 x0 = a0[0], x1 = a0[1], y0 = a1[0], y1 = a1[1];
    v16bf av;
    av[0]  = (__bf16)(x0.x * ascale); av[1]  = (__bf16)(x0.y * ascale);
    av[2]  = (__bf16)(x0.z * ascale); av[3]  = (__bf16)(x0.w * ascale);
    av[4]  = (__bf16)(x1.x * ascale); av[5]  = (__bf16)(x1.y * ascale);
    av[6]  = (__bf16)(x1.z * ascale); av[7]  = (__bf16)(x1.w * ascale);
    av[8]  = (__bf16)(y0.x * ascale); av[9]  = (__bf16)(y0.y * ascale);
    av[10] = (__bf16)(y0.z * ascale); av[11] = (__bf16)(y0.w * ascale);
    av[12] = (__bf16)(y1.x * ascale); av[13] = (__bf16)(y1.y * ascale);
    av[14] = (__bf16)(y1.z * ascale); av[15] = (__bf16)(y1.w * ascale);
    const v16bf bv = bvp[ks * 32];
    acc = __builtin_amdgcn_wmma_f32_16x16x32_bf16(
        false, av, false, bv, (short)0, acc, false, false);
  }

  // C/D layout: VGPR v -> M = m0 + half*8 + v, N = n0 + l16
  const int n = n0 + l16;
  const float bval = HASBIAS ? bias[n] : 0.0f;
#pragma unroll
  for (int v = 0; v < 8; ++v) {
    float val = acc[v] + bval;
    if (RELU) val = fmaxf(val, 0.0f);
    C[(size_t)(m0 + half * 8 + v) * Ncols + n] = val;
  }
}

// ---------------------------------------------------------------------------
// Per-(node,head) attention logits: aS = <xl, att_src>, aD = <xl, att_dst>.
// ---------------------------------------------------------------------------
__global__ void k_att(const float* __restrict__ xl, const float* __restrict__ att_src,
                      const float* __restrict__ att_dst,
                      float* __restrict__ aS, float* __restrict__ aD, int N) {
  int t = blockIdx.x * blockDim.x + threadIdx.x;
  if (t >= N * NHEADS) return;
  const int node = t >> 2, hd = t & 3;
  const float* __restrict__ xp = xl + (size_t)node * HIDDEN + hd * CH;
  const float* __restrict__ as = att_src + hd * CH;
  const float* __restrict__ ad = att_dst + hd * CH;
  float s = 0.f, d = 0.f;
#pragma unroll
  for (int c = 0; c < CH; ++c) { float v = xp[c]; s += v * as[c]; d += v * ad[c]; }
  aS[t] = s; aD[t] = d;
}

// ---------------------------------------------------------------------------
// Per-layer scratch init (atomics accumulate => must re-init every call).
// ---------------------------------------------------------------------------
__global__ void k_init_layer(float* __restrict__ agg, unsigned* __restrict__ maxb,
                             float* __restrict__ denom, int N) {
  int t = blockIdx.x * blockDim.x + threadIdx.x;
  if (t < N * HIDDEN) agg[t] = 0.0f;
  if (t < N * NHEADS) { maxb[t] = ENC_NEG_INF; denom[t] = 0.0f; }
}

__global__ void k_init_scal(float* scal) {
  if (threadIdx.x == 0) { scal[0] = 0.0f; scal[1] = 0.0f; }
}

// ---------------------------------------------------------------------------
// Edge pass 1: scatter-max of leaky_relu(aS[src]+aD[dst]) into dst buckets.
// ---------------------------------------------------------------------------
__global__ void k_edge_max(const int* __restrict__ src, const int* __restrict__ dst,
                           const float* __restrict__ aS, const float* __restrict__ aD,
                           unsigned* __restrict__ maxb, int E, int Etot) {
  int t = blockIdx.x * blockDim.x + threadIdx.x;
  if (t >= Etot * NHEADS) return;
  const int e = t >> 2, hd = t & 3;
  const int s = (e < E) ? src[e] : (e - E);
  const int d = (e < E) ? dst[e] : (e - E);
  float v = aS[s * NHEADS + hd] + aD[d * NHEADS + hd];
  v = (v > 0.0f) ? v : NEG_SLOPE * v;
  atomicMax(&maxb[d * NHEADS + hd], encOrd(v));
}

// ---------------------------------------------------------------------------
// Edge pass 2: ex = exp(e - max[dst]); store ex, scatter-add denominators.
// ---------------------------------------------------------------------------
__global__ void k_edge_exp(const int* __restrict__ src, const int* __restrict__ dst,
                           const float* __restrict__ aS, const float* __restrict__ aD,
                           const unsigned* __restrict__ maxb,
                           float* __restrict__ exbuf, float* __restrict__ denom,
                           int E, int Etot) {
  int t = blockIdx.x * blockDim.x + threadIdx.x;
  if (t >= Etot * NHEADS) return;
  const int e = t >> 2, hd = t & 3;
  const int s = (e < E) ? src[e] : (e - E);
  const int d = (e < E) ? dst[e] : (e - E);
  float v = aS[s * NHEADS + hd] + aD[d * NHEADS + hd];
  v = (v > 0.0f) ? v : NEG_SLOPE * v;
  const float ex = __expf(v - decOrd(maxb[d * NHEADS + hd]));
  exbuf[t] = ex;
  atomicAdd(&denom[d * NHEADS + hd], ex);
}

// ---------------------------------------------------------------------------
// Edge pass 3: agg[dst,c] += xl[src,c] * alpha(e,head). One thread per
// (edge, channel); ~109M f32 atomics resolving mostly in the 192MB L2.
// ---------------------------------------------------------------------------
__global__ void k_edge_scatter(const int* __restrict__ src, const int* __restrict__ dst,
                               const float* __restrict__ xl,
                               const float* __restrict__ exbuf,
                               const float* __restrict__ denom,
                               float* __restrict__ agg, int E, int Etot) {
  long long t = (long long)blockIdx.x * blockDim.x + threadIdx.x;
  if (t >= (long long)Etot * HIDDEN) return;
  const int e  = (int)(t >> 7);
  const int c  = (int)(t & (HIDDEN - 1));
  const int hd = c >> 5;
  const int s = (e < E) ? src[e] : (e - E);
  const int d = (e < E) ? dst[e] : (e - E);
  const float alpha = exbuf[(size_t)e * NHEADS + hd] /
                      (denom[d * NHEADS + hd] + 1e-16f);
  atomicAdd(&agg[(size_t)d * HIDDEN + c], xl[(size_t)s * HIDDEN + c] * alpha);
}

// ---------------------------------------------------------------------------
// Fused residual + bias + LayerNorm over 128 channels. One wave32 per row,
// 4 values per lane, __shfl_xor reductions. Compile-time residual/bias.
// ---------------------------------------------------------------------------
template <bool HASRES, bool HASBIAS>
__global__ void k_ln(const float* __restrict__ x, const float* __restrict__ res,
                     const float* __restrict__ bias, const float* __restrict__ g,
                     const float* __restrict__ b, float* __restrict__ out, int N) {
  const int wave = threadIdx.x >> 5;
  const int lane = threadIdx.x & 31;
  const int row = blockIdx.x * (blockDim.x >> 5) + wave;
  if (row >= N) return;
  const float* __restrict__ xr = x + (size_t)row * HIDDEN;
  const float* __restrict__ rr = res + (size_t)row * HIDDEN;
  float v[4];
  float s = 0.0f;
#pragma unroll
  for (int j = 0; j < 4; ++j) {
    const int c = lane + 32 * j;
    float t = xr[c];
    if (HASRES)  t += rr[c];
    if (HASBIAS) t += bias[c];
    v[j] = t; s += t;
  }
#pragma unroll
  for (int o = 16; o > 0; o >>= 1) s += __shfl_xor(s, o, 32);
  const float mu = s * (1.0f / HIDDEN);
  float var = 0.0f;
#pragma unroll
  for (int j = 0; j < 4; ++j) { float dlt = v[j] - mu; var += dlt * dlt; }
#pragma unroll
  for (int o = 16; o > 0; o >>= 1) var += __shfl_xor(var, o, 32);
  const float inv = rsqrtf(var * (1.0f / HIDDEN) + 1e-5f);
#pragma unroll
  for (int j = 0; j < 4; ++j) {
    const int c = lane + 32 * j;
    out[(size_t)row * HIDDEN + c] = (v[j] - mu) * inv * g[c] + b[c];
  }
}

// ---------------------------------------------------------------------------
// Masked MSE pieces.
// ---------------------------------------------------------------------------
__global__ void k_loss_sq(const float* __restrict__ recon, const float* __restrict__ x,
                          const int* __restrict__ mask, float* __restrict__ scal,
                          long long total) {
  __shared__ float sh[256];
  long long t = (long long)blockIdx.x * blockDim.x + threadIdx.x;
  float val = 0.0f;
  if (t < total) {
    const int node = (int)(t >> 6);   // DIN == 64
    if (mask[node]) { float d = recon[t] - x[t]; val = d * d; }
  }
  sh[threadIdx.x] = val;
  __syncthreads();
  for (int s = 128; s > 0; s >>= 1) {
    if (threadIdx.x < s) sh[threadIdx.x] += sh[threadIdx.x + s];
    __syncthreads();
  }
  if (threadIdx.x == 0) atomicAdd(&scal[0], sh[0]);
}

__global__ void k_count(const int* __restrict__ mask, float* __restrict__ scal, int N) {
  __shared__ float sh[256];
  int t = blockIdx.x * blockDim.x + threadIdx.x;
  sh[threadIdx.x] = (t < N && mask[t]) ? 1.0f : 0.0f;
  __syncthreads();
  for (int s = 128; s > 0; s >>= 1) {
    if (threadIdx.x < s) sh[threadIdx.x] += sh[threadIdx.x + s];
    __syncthreads();
  }
  if (threadIdx.x == 0) atomicAdd(&scal[1], sh[0]);
}

__global__ void k_final(const float* __restrict__ scal, float* __restrict__ out) {
  if (threadIdx.x == 0)
    out[0] = scal[0] / (fmaxf(scal[1], 1.0f) * (float)DIN);
}

// ---------------------------------------------------------------------------
// Host-side orchestration (all on `stream`, graph-capture safe).
// ---------------------------------------------------------------------------
extern "C" void kernel_launch(void* const* d_in, const int* in_sizes, int n_in,
                              void* d_out, int out_size, void* d_ws, size_t ws_size,
                              hipStream_t stream) {
  const float* x    = (const float*)d_in[0];
  const int*   ei   = (const int*)d_in[1];
  const int*   mask = (const int*)d_in[2];
  const float* W_in = (const float*)d_in[3];
  const float* b_in = (const float*)d_in[4];
  const float* dec_W1   = (const float*)d_in[17];
  const float* dec_b1   = (const float*)d_in[18];
  const float* dec_ln_g = (const float*)d_in[19];
  const float* dec_ln_b = (const float*)d_in[20];
  const float* dec_W2   = (const float*)d_in[21];
  const float* dec_b2   = (const float*)d_in[22];

  const int N = in_sizes[2];          // 50000
  const int E = in_sizes[1] / 2;      // 800000
  const int Etot = E + N;             // self-loops appended
  const int* src = ei;                // edge_index[0]
  const int* dst = ei + E;            // edge_index[1]

  float* recon = (float*)d_out + 1;   // d_out = [loss, recon(N*DIN)]

  // Workspace layout (floats first, then 32B-aligned bf16 weight buffers).
  float*    h0    = (float*)d_ws;
  float*    h1    = h0  + (size_t)N * HIDDEN;
  float*    xl    = h1  + (size_t)N * HIDDEN;
  float*    agg   = xl  + (size_t)N * HIDDEN;
  float*    aS    = agg + (size_t)N * HIDDEN;
  float*    aD    = aS  + (size_t)N * NHEADS;
  unsigned* maxb  = (unsigned*)(aD + (size_t)N * NHEADS);
  float*    denom = (float*)(maxb + (size_t)N * NHEADS);
  float*    exbuf = denom + (size_t)N * NHEADS;
  float*    scal  = exbuf + (size_t)Etot * NHEADS;
  __bf16*   wr_in = (__bf16*)(scal + 8);            // 8-float pad keeps 32B align
  __bf16*   wr_g0 = wr_in + (size_t)DIN * HIDDEN;
  __bf16*   wr_g1 = wr_g0 + (size_t)HIDDEN * HIDDEN;
  __bf16*   wr_d1 = wr_g1 + (size_t)HIDDEN * HIDDEN;
  __bf16*   wr_d2 = wr_d1 + (size_t)HIDDEN * HIDDEN;

  const int mtiles = (N + 15) / 16;
  const dim3 blk256(256), blk128(128);
  const int gNH   = (N * NHEADS + 255) / 256;
  const int gEH   = (Etot * NHEADS + 255) / 256;
  const long long scat_total = (long long)Etot * HIDDEN;
  const int gScat = (int)((scat_total + 255) / 256);
  const int gRows = (N + 7) / 8;                    // 8 rows (waves) per block

  // 0) repack all weight matrices to bf16 WMMA lane layout (cheap, <= 64KB each)
  k_repack_w<<<(DIN * HIDDEN / 16 + 255) / 256, blk256, 0, stream>>>(W_in, wr_in, DIN, HIDDEN);
  k_repack_w<<<(HIDDEN * HIDDEN / 16 + 255) / 256, blk256, 0, stream>>>(
      (const float*)d_in[5], wr_g0, HIDDEN, HIDDEN);
  k_repack_w<<<(HIDDEN * HIDDEN / 16 + 255) / 256, blk256, 0, stream>>>(
      (const float*)d_in[11], wr_g1, HIDDEN, HIDDEN);
  k_repack_w<<<(HIDDEN * HIDDEN / 16 + 255) / 256, blk256, 0, stream>>>(dec_W1, wr_d1, HIDDEN, HIDDEN);
  k_repack_w<<<(HIDDEN * DIN / 16 + 255) / 256, blk256, 0, stream>>>(dec_W2, wr_d2, HIDDEN, DIN);

  // 1) masked input projection: h0 = mask(x) @ W_in + b_in  (WMMA bf16)
  k_gemm_bf16<DIN, true, false, true><<<mtiles, blk256, 0, stream>>>(
      x, wr_in, b_in, mask, h0, N, HIDDEN);

  // 2) two GAT layers (ping-pong h0 <-> h1)
  float* hin = h0;
  float* hout = h1;
  for (int layer = 0; layer < 2; ++layer) {
    const int pb = 5 + 6 * layer;
    const __bf16* wr       = (layer == 0) ? wr_g0 : wr_g1;
    const float* att_src  = (const float*)d_in[pb + 1];
    const float* att_dst  = (const float*)d_in[pb + 2];
    const float* gat_bias = (const float*)d_in[pb + 3];
    const float* ln_g     = (const float*)d_in[pb + 4];
    const float* ln_b     = (const float*)d_in[pb + 5];

    k_gemm_bf16<HIDDEN, false, false, false><<<mtiles, blk256, 0, stream>>>(
        hin, wr, nullptr, nullptr, xl, N, HIDDEN);
    k_att<<<gNH, blk256, 0, stream>>>(xl, att_src, att_dst, aS, aD, N);
    k_init_layer<<<(N * HIDDEN + 255) / 256, blk256, 0, stream>>>(agg, maxb, denom, N);
    k_edge_max<<<gEH, blk256, 0, stream>>>(src, dst, aS, aD, maxb, E, Etot);
    k_edge_exp<<<gEH, blk256, 0, stream>>>(src, dst, aS, aD, maxb, exbuf, denom, E, Etot);
    k_edge_scatter<<<gScat, blk256, 0, stream>>>(src, dst, xl, exbuf, denom, agg, E, Etot);
    k_ln<true, true><<<gRows, blk256, 0, stream>>>(hin, agg, gat_bias, ln_g, ln_b, hout, N);

    float* t = hin; hin = hout; hout = t;           // hin now holds layer output
  }
  // after 2 swaps, final encoder output is in h0 (== hin)

  // 3) decoder: relu(h @ W1 + b1) -> LN -> @ W2 + b2
  k_gemm_bf16<HIDDEN, true, true, false><<<mtiles, blk256, 0, stream>>>(
      hin, wr_d1, dec_b1, nullptr, xl, N, HIDDEN);
  k_ln<false, false><<<gRows, blk256, 0, stream>>>(xl, nullptr, nullptr,
                                                   dec_ln_g, dec_ln_b, agg, N);
  k_gemm_bf16<HIDDEN, true, false, false><<<mtiles, blk128, 0, stream>>>(
      agg, wr_d2, dec_b2, nullptr, recon, N, DIN);

  // 4) masked MSE
  k_init_scal<<<1, 32, 0, stream>>>(scal);
  const long long ltotal = (long long)N * DIN;
  k_loss_sq<<<(int)((ltotal + 255) / 256), blk256, 0, stream>>>(recon, x, mask, scal, ltotal);
  k_count<<<(N + 255) / 256, blk256, 0, stream>>>(mask, scal, N);
  k_final<<<1, 32, 0, stream>>>(scal, (float*)d_out);
}
